// QuantAttentionPool2d_78348793414257
// MI455X (gfx1250) — matrix operand order repaired and checked by reading, not compiled, not run on hardware
//
#include <hip/hip_runtime.h>

// ---------------------------------------------------------------------------
// QuantAttentionPool2d for MI455X (gfx1250, wave32, WMMA + async LDS staging)
// N=512, C=2048, L=50 (1 mean token + 49 spatial), H=32 heads, hd=64, OUT=1024
// ---------------------------------------------------------------------------

typedef __attribute__((ext_vector_type(16))) __bf16 v16bf;
typedef __attribute__((ext_vector_type(8)))  __bf16 v8bf;
typedef __attribute__((ext_vector_type(8)))  float  v8f;
typedef __attribute__((ext_vector_type(4)))  int    v4i;

#define NB   512
#define CCH  2048
#define LTOK 50
#define NH   32
#define HD   64
#define COUT 1024

#define KSTEP 32          // K per pipeline stage (= WMMA K)
#define SSTR  40          // LDS row stride in bf16 elems (32 + 8 pad -> 80B)

#if defined(__has_builtin)
#  if __has_builtin(__builtin_amdgcn_global_load_async_to_lds_b128)
#    define HAVE_ASYNC_BUILTIN 1
#  endif
#  if __has_builtin(__builtin_amdgcn_s_wait_asynccnt)
#    define HAVE_WAIT_ASYNC_BUILTIN 1
#  endif
#endif

// builtin signature (from clang diagnostic): (v4i AS1* src, v4i AS3* dst, Ii, Ii)
typedef __attribute__((address_space(1))) v4i gv4i;
typedef __attribute__((address_space(3))) v4i lv4i;
typedef __attribute__((address_space(3))) void as3_void;

// per-lane 16B global -> LDS async copy (tracked by ASYNCcnt)
__device__ __forceinline__ void async_ld_16B(const void* gsrc, void* ldst) {
#if defined(HAVE_ASYNC_BUILTIN)
    __builtin_amdgcn_global_load_async_to_lds_b128((gv4i*)gsrc, (lv4i*)ldst, 0, 0);
#else
    unsigned loff = (unsigned)(unsigned long long)(as3_void*)ldst;
    asm volatile("global_load_async_to_lds_b128 %0, %1, off"
                 :: "v"(loff), "v"((unsigned long long)gsrc) : "memory");
#endif
}

template <int N>
__device__ __forceinline__ void wait_async() {
#if defined(HAVE_WAIT_ASYNC_BUILTIN)
    __builtin_amdgcn_s_wait_asynccnt(N);
#else
    if (N == 0) asm volatile("s_wait_asynccnt 0x0" ::: "memory");
    else        asm volatile("s_wait_asynccnt 0x4" ::: "memory");
#endif
}

// fake-quant to int8 grid; s is precomputed scale (max/127 + 1e-8)
__device__ __forceinline__ float fq(float x, float s) {
    float r = rintf(x / s);
    r = fminf(fmaxf(r, -128.f), 127.f);
    return r * s;
}

// ---------------------------------------------------------------------------
__global__ void init_scales_kernel(unsigned* s) {
    if (threadIdx.x < 8) s[threadIdx.x] = 0u;   // +0.0f bit pattern
}

__global__ void maxabs_kernel(const float* __restrict__ w, long n, unsigned* slot) {
    __shared__ float red[256];
    long i = (long)blockIdx.x * blockDim.x + threadIdx.x;
    float m = 0.f;
    for (; i < n; i += (long)gridDim.x * blockDim.x) m = fmaxf(m, fabsf(w[i]));
    red[threadIdx.x] = m;
    __syncthreads();
    for (int s2 = 128; s2 > 0; s2 >>= 1) {
        if ((int)threadIdx.x < s2) red[threadIdx.x] = fmaxf(red[threadIdx.x], red[threadIdx.x + s2]);
        __syncthreads();
    }
    if (threadIdx.x == 0) atomicMax(slot, __float_as_uint(red[0]));
}

__global__ void quant_weight_kernel(const float* __restrict__ w, long n,
                                    const float* __restrict__ slot, __bf16* __restrict__ dst) {
    float s = slot[0] / 127.f + 1e-8f;
    long i = (long)blockIdx.x * blockDim.x + threadIdx.x;
    for (; i < n; i += (long)gridDim.x * blockDim.x) {
        float r = rintf(w[i] / s);
        r = fminf(fmaxf(r, -128.f), 127.f);
        dst[i] = (__bf16)(r * s);
    }
}

// Build T[n, l, c] = (l==0 ? mean_s x[n,c,s] : x[n,c,l-1]) + pos[l,c]   (bf16)
__global__ void build_tokens_kernel(const float* __restrict__ x,
                                    const float* __restrict__ pos,
                                    __bf16* __restrict__ T) {
    long idx = (long)blockIdx.x * blockDim.x + threadIdx.x;   // 0 .. N*C-1
    int n = (int)(idx / CCH);
    int c = (int)(idx % CCH);
    const float* xp = x + ((long)n * CCH + c) * 49;
    float sum = 0.f;
    #pragma unroll 7
    for (int s = 0; s < 49; ++s) sum += xp[s];
    float mean = sum * (1.f / 49.f);
    __bf16* tp = T + (long)n * LTOK * CCH + c;
    tp[0] = (__bf16)(mean + pos[c]);
    for (int l = 1; l < LTOK; ++l)
        tp[(long)l * CCH] = (__bf16)(xp[l - 1] + pos[(long)l * CCH + c]);
}

// ---------------------------------------------------------------------------
// bf16 WMMA GEMM:  C[m,n] = sum_k A[m,k]*B[n,k] + bias[n]
// Block = 256 thr (8 waves), block tile 128x128, wave tile 64x32 (4x2 WMMA).
// Double-buffered LDS staging fed by global_load_async_to_lds_b128; compute
// reads fragments from LDS (ds_load_b128) while the next stage is in flight.
// All dims must be multiples of the tiling (true for every call here).
// ---------------------------------------------------------------------------
template <bool OUTF32>
__global__ void gemm_bf16_wmma(const __bf16* __restrict__ A, long lda,
                               const __bf16* __restrict__ B, long ldb,
                               const float* __restrict__ bias,
                               void* __restrict__ Cout, long ldc,
                               int Kdim, unsigned* maxslot) {
    __shared__ alignas(16) __bf16 sA[2][128 * SSTR];
    __shared__ alignas(16) __bf16 sB[2][128 * SSTR];
    __shared__ float red[256];

    const int tid  = threadIdx.x;
    const int lane = tid & 31;
    const int wave = tid >> 5;
    const int wm = wave >> 2;           // 0..1
    const int wn = wave & 3;            // 0..3
    const int row16 = lane & 15;
    const int khi   = lane >> 4;        // 0 or 1
    const long blockM0 = (long)blockIdx.y * 128;
    const long blockN0 = (long)blockIdx.x * 128;

    // Stage copy: 128 rows x 32 halves (64B = 4x16B chunks) per matrix.
    // 512 transfers per matrix / 256 threads -> 2 each (4 async instr/wave).
    const int i0 = tid, i1 = tid + 256;
    const int r0 = i0 >> 2, c0 = (i0 & 3) * 8;
    const int r1 = i1 >> 2, c1 = (i1 & 3) * 8;

    auto issue = [&](int buf, int k0) {
        async_ld_16B(A + (blockM0 + r0) * lda + k0 + c0, &sA[buf][r0 * SSTR + c0]);
        async_ld_16B(A + (blockM0 + r1) * lda + k0 + c1, &sA[buf][r1 * SSTR + c1]);
        async_ld_16B(B + (blockN0 + r0) * ldb + k0 + c0, &sB[buf][r0 * SSTR + c0]);
        async_ld_16B(B + (blockN0 + r1) * ldb + k0 + c1, &sB[buf][r1 * SSTR + c1]);
    };

    v8f acc[4][2];
    const v8f vzero = {0.f, 0.f, 0.f, 0.f, 0.f, 0.f, 0.f, 0.f};
    #pragma unroll
    for (int mt = 0; mt < 4; ++mt)
        #pragma unroll
        for (int nt = 0; nt < 2; ++nt) acc[mt][nt] = vzero;

    const int nst = Kdim / KSTEP;
    issue(0, 0);
    if (nst > 1) issue(1, KSTEP);

    for (int s = 0; s < nst; ++s) {
        // oldest stage (4 async ops) complete; async loads retire in order.
        if (s + 1 < nst) wait_async<4>();
        else             wait_async<0>();
        __syncthreads();

        const int buf = s & 1;
        v16bf a[4], b[2];
        // A fragment 16x32: lane holds row (lane&15); K interleave per ISA:
        // lanes 0-15 -> K 0-7,16-23 ; lanes 16-31 -> K 8-15,24-31
        #pragma unroll
        for (int mt = 0; mt < 4; ++mt) {
            const __bf16* p = &sA[buf][(wm * 64 + mt * 16 + row16) * SSTR];
            v8bf lo = *(const v8bf*)(p + khi * 8);
            v8bf hi = *(const v8bf*)(p + 16 + khi * 8);
            #pragma unroll
            for (int i = 0; i < 8; ++i) { a[mt][i] = lo[i]; a[mt][8 + i] = hi[i]; }
        }
        // B fragment 32x16: lane holds column; lanes 0-15 K 0-7,16-23 ; lanes 16-31 K 8-15,24-31
        #pragma unroll
        for (int nt = 0; nt < 2; ++nt) {
            const __bf16* p = &sB[buf][(wn * 32 + nt * 16 + row16) * SSTR];
            v8bf lo = *(const v8bf*)(p + khi * 8);
            v8bf hi = *(const v8bf*)(p + 16 + khi * 8);
            #pragma unroll
            for (int i = 0; i < 8; ++i) { b[nt][i] = lo[i]; b[nt][8 + i] = hi[i]; }
        }
        #pragma unroll
        for (int mt = 0; mt < 4; ++mt)
            #pragma unroll
            for (int nt = 0; nt < 2; ++nt)
                acc[mt][nt] = __builtin_amdgcn_wmma_f32_16x16x32_bf16(
                    false, a[mt], false, b[nt], (short)0, acc[mt][nt], false, false);

        __syncthreads();                       // all waves done reading buf
        if (s + 2 < nst) issue(buf, (s + 2) * KSTEP);
    }

    // Epilogue: D layout — VGPR r: lanes 0-15 (M=r, N=lane), lanes 16-31 (M=8+r)
    float wmaxv = 0.f;
    #pragma unroll
    for (int mt = 0; mt < 4; ++mt)
        #pragma unroll
        for (int nt = 0; nt < 2; ++nt)
            #pragma unroll
            for (int r = 0; r < 8; ++r) {
                long m = blockM0 + wm * 64 + mt * 16 + khi * 8 + r;
                long n = blockN0 + wn * 32 + nt * 16 + row16;
                float val = acc[mt][nt][r] + bias[n];
                wmaxv = fmaxf(wmaxv, fabsf(val));
                if (OUTF32) ((float*)Cout)[m * ldc + n] = val;
                else        ((__bf16*)Cout)[m * ldc + n] = (__bf16)val;
            }

    if (maxslot) {
        red[tid] = wmaxv;
        __syncthreads();
        for (int s2 = 128; s2 > 0; s2 >>= 1) {
            if (tid < s2) red[tid] = fmaxf(red[tid], red[tid + s2]);
            __syncthreads();
        }
        if (tid == 0) atomicMax(maxslot, __float_as_uint(red[0]));
    }
}

// ---------------------------------------------------------------------------
// Attention core: per (n,h) block of 64 threads.
// logits = fq(q) . fq(k) / 8 ; softmax over 50 ; store probs ; record prob max
// ---------------------------------------------------------------------------
__global__ void attn_kernel(const __bf16* __restrict__ Qa, const __bf16* __restrict__ Ka,
                            const float* __restrict__ scales, float* __restrict__ probs,
                            unsigned* amax_slot) {
    __shared__ float qs[64];
    __shared__ float red[64];
    const int nh = blockIdx.x;
    const int n = nh >> 5, h = nh & 31;
    const int t = threadIdx.x;

    const float sq = scales[4] / 127.f + 1e-8f;
    const float sk = scales[5] / 127.f + 1e-8f;

    qs[t] = fq((float)Qa[(long)n * CCH + h * HD + t], sq);
    __syncthreads();

    float logit = -1e30f;
    if (t < LTOK) {
        const __bf16* kr = Ka + ((long)(n * LTOK + t)) * CCH + h * HD;
        float acc = 0.f;
        #pragma unroll 8
        for (int d = 0; d < HD; ++d) acc += qs[d] * fq((float)kr[d], sk);
        logit = acc * 0.125f;   // 1/sqrt(64)
    }
    red[t] = logit;
    __syncthreads();
    for (int s2 = 32; s2 > 0; s2 >>= 1) {
        if (t < s2) red[t] = fmaxf(red[t], red[t + s2]);
        __syncthreads();
    }
    const float mx = red[0];
    __syncthreads();

    const float e = (t < LTOK) ? __expf(logit - mx) : 0.f;
    red[t] = e;
    __syncthreads();
    for (int s2 = 32; s2 > 0; s2 >>= 1) {
        if (t < s2) red[t] += red[t + s2];
        __syncthreads();
    }
    const float denom = red[0];
    __syncthreads();

    const float p = e / denom;
    if (t < LTOK) probs[(long)nh * LTOK + t] = p;

    red[t] = p;
    __syncthreads();
    for (int s2 = 32; s2 > 0; s2 >>= 1) {
        if (t < s2) red[t] = fmaxf(red[t], red[t + s2]);
        __syncthreads();
    }
    if (t == 0) atomicMax(amax_slot, __float_as_uint(red[0]));
}

// o[n, h*64+d] = sum_l fq(prob) * fq(v)
__global__ void av_kernel(const float* __restrict__ probs, const __bf16* __restrict__ Va,
                          const float* __restrict__ scales, __bf16* __restrict__ O) {
    const int nh = blockIdx.x;
    const int n = nh >> 5, h = nh & 31;
    const int d = threadIdx.x;
    const float sv = scales[6] / 127.f + 1e-8f;
    const float sa = scales[7] / 127.f + 1e-8f;
    float acc = 0.f;
    #pragma unroll 5
    for (int l = 0; l < LTOK; ++l) {
        float p = fq(probs[(long)nh * LTOK + l], sa);
        acc += p * fq((float)Va[((long)(n * LTOK + l)) * CCH + h * HD + d], sv);
    }
    O[(long)n * CCH + h * HD + d] = (__bf16)acc;
}

// ---------------------------------------------------------------------------
extern "C" void kernel_launch(void* const* d_in, const int* in_sizes, int n_in,
                              void* d_out, int out_size, void* d_ws, size_t ws_size,
                              hipStream_t stream) {
    (void)in_sizes; (void)n_in; (void)out_size; (void)ws_size;
    const float* x   = (const float*)d_in[0];
    const float* pos = (const float*)d_in[1];
    const float* wq  = (const float*)d_in[2];
    const float* bq  = (const float*)d_in[3];
    const float* wk  = (const float*)d_in[4];
    const float* bk  = (const float*)d_in[5];
    const float* wv  = (const float*)d_in[6];
    const float* bv  = (const float*)d_in[7];
    const float* wc  = (const float*)d_in[8];
    const float* bc  = (const float*)d_in[9];
    float* out = (float*)d_out;

    // workspace layout
    // slots: [0..3] weight maxabs (wq,wk,wv,wc) [4] qmax [5] kmax [6] vmax [7] prob max
    unsigned* scales_u = (unsigned*)d_ws;
    const float* scales_f = (const float*)d_ws;
    char* p = (char*)d_ws + 256;
    __bf16* wqq = (__bf16*)p; p += (size_t)CCH * CCH * 2;
    __bf16* wkq = (__bf16*)p; p += (size_t)CCH * CCH * 2;
    __bf16* wvq = (__bf16*)p; p += (size_t)CCH * CCH * 2;
    __bf16* wcq = (__bf16*)p; p += (size_t)COUT * CCH * 2;
    __bf16* T   = (__bf16*)p; p += (size_t)NB * LTOK * CCH * 2;
    __bf16* Ka  = (__bf16*)p; p += (size_t)NB * LTOK * CCH * 2;
    __bf16* Va  = (__bf16*)p; p += (size_t)NB * LTOK * CCH * 2;
    __bf16* Qa  = (__bf16*)p; p += (size_t)NB * CCH * 2;
    __bf16* O   = (__bf16*)p; p += (size_t)NB * CCH * 2;
    float*  probs = (float*)p;

    init_scales_kernel<<<1, 32, 0, stream>>>(scales_u);

    maxabs_kernel<<<512, 256, 0, stream>>>(wq, (long)CCH * CCH, scales_u + 0);
    maxabs_kernel<<<512, 256, 0, stream>>>(wk, (long)CCH * CCH, scales_u + 1);
    maxabs_kernel<<<512, 256, 0, stream>>>(wv, (long)CCH * CCH, scales_u + 2);
    maxabs_kernel<<<512, 256, 0, stream>>>(wc, (long)COUT * CCH, scales_u + 3);

    quant_weight_kernel<<<512, 256, 0, stream>>>(wq, (long)CCH * CCH, scales_f + 0, wqq);
    quant_weight_kernel<<<512, 256, 0, stream>>>(wk, (long)CCH * CCH, scales_f + 1, wkq);
    quant_weight_kernel<<<512, 256, 0, stream>>>(wv, (long)CCH * CCH, scales_f + 2, wvq);
    quant_weight_kernel<<<512, 256, 0, stream>>>(wc, (long)COUT * CCH, scales_f + 3, wcq);

    build_tokens_kernel<<<(NB * CCH) / 256, 256, 0, stream>>>(x, pos, T);

    // K = T @ fq(wk)^T + bk     M=25600 N=2048 K=2048
    gemm_bf16_wmma<false><<<dim3(CCH / 128, (NB * LTOK) / 128), 256, 0, stream>>>(
        T, CCH, wkq, CCH, bk, (void*)Ka, CCH, CCH, scales_u + 5);
    // V = T @ fq(wv)^T + bv
    gemm_bf16_wmma<false><<<dim3(CCH / 128, (NB * LTOK) / 128), 256, 0, stream>>>(
        T, CCH, wvq, CCH, bv, (void*)Va, CCH, CCH, scales_u + 6);
    // Q = T[:,0,:] @ fq(wq)^T + bq   (mean-token rows, lda = L*C)   M=512
    gemm_bf16_wmma<false><<<dim3(CCH / 128, NB / 128), 256, 0, stream>>>(
        T, (long)LTOK * CCH, wqq, CCH, bq, (void*)Qa, CCH, CCH, scales_u + 4);

    attn_kernel<<<NB * NH, 64, 0, stream>>>(Qa, Ka, scales_f, probs, scales_u + 7);
    av_kernel<<<NB * NH, 64, 0, stream>>>(probs, Va, scales_f, O);

    // out = O @ fq(wc)^T + bc   M=512 N=1024 K=2048, fp32 output
    gemm_bf16_wmma<true><<<dim3(COUT / 128, NB / 128), 256, 0, stream>>>(
        O, CCH, wcq, CCH, bc, (void*)out, COUT, CCH, nullptr);
}
